// TransformerDecoderLayer_69329362092329
// MI455X (gfx1250) — matrix operand order repaired and checked by reading, not compiled, not run on hardware
//
#include <hip/hip_runtime.h>
#include <cstdint>
#include <cstddef>

typedef _Float16 half_t;
typedef __attribute__((ext_vector_type(8)))  half_t v8h;
typedef __attribute__((ext_vector_type(16))) half_t v16h;
typedef __attribute__((ext_vector_type(8)))  float  v8f;

// ---- problem constants -------------------------------------------------
static constexpr int kB   = 8;
static constexpr int kN   = 900;          // queries
static constexpr int kD   = 256;
static constexpr int kH   = 8;            // heads (SA_H == CA_H)
static constexpr int kDH  = 32;           // head dim
static constexpr int kL   = 4;
static constexpr int kP   = 4;
static constexpr int kFF  = 2048;
static constexpr int kS   = 21760;        // memory tokens
static constexpr int kRows = kB * kN;     // 7200
static constexpr int kMemRows = kB * kS;  // 174080

__device__ __forceinline__ v16h cat8(v8h lo, v8h hi) {
  v16h r;
#pragma unroll
  for (int i = 0; i < 8; ++i) { r[i] = lo[i]; r[i + 8] = hi[i]; }
  return r;
}
__device__ __forceinline__ v16h pack16(const half_t* t) {
  v16h r;
#pragma unroll
  for (int i = 0; i < 16; ++i) r[i] = t[i];
  return r;
}

// LDS byte-offset of a generic pointer to __shared__ (addrspacecast -> AS3, 32-bit)
typedef __attribute__((address_space(3))) half_t lds_half_t;
__device__ __forceinline__ uint32_t lds_off_of(const half_t* p) {
  return (uint32_t)(uintptr_t)(lds_half_t*)p;
}
// CDNA5 async global->LDS copy of 16 bytes (tracked by ASYNCcnt)
__device__ __forceinline__ void async_copy_b128(const half_t* gsrc, const half_t* ldst) {
  const uint64_t ga = (uint64_t)(uintptr_t)gsrc;
  const uint32_t la = lds_off_of(ldst);
  asm volatile("global_load_async_to_lds_b128 %0, %1, off"
               :: "v"(la), "v"(ga) : "memory");
}
__device__ __forceinline__ void wait_asynccnt0() {
  asm volatile("s_wait_asynccnt 0x0" ::: "memory");
}

// =======================================================================
// RoPE prep:  qk = rope(tgt + query_pos), tgt_h = f16(tgt)
// grid = B*N blocks, 128 threads (one f16 pair per thread)
// =======================================================================
__global__ __launch_bounds__(128) void rope_kernel(
    const float* __restrict__ tgt, const float* __restrict__ qpos,
    const float* __restrict__ boxes,
    half_t* __restrict__ qk_h, half_t* __restrict__ tgt_h)
{
  const int bn = blockIdx.x;
  const int pr = threadIdx.x;        // pair index 0..127
  const int d0 = pr * 2, d1 = d0 + 1;
  const float* t = tgt  + (size_t)bn * kD;
  const float* q = qpos + (size_t)bn * kD;
  const float t0 = t[d0], t1 = t[d1];
  tgt_h[(size_t)bn * kD + d0] = (half_t)t0;
  tgt_h[(size_t)bn * kD + d1] = (half_t)t1;
  const float a0 = t0 + q[d0];
  const float a1 = t1 + q[d1];
  float r0 = a0, r1 = a1;
  if (d0 < 252) {                    // D_ROPE = 3 * 84
    const int band   = d0 / 84;
    const int within = d0 - band * 84;
    const int fi     = within >> 1;  // 0..41
    float f = __powf(10000.0f, -(float)fi / 42.0f);
    const float cx = boxes[(size_t)bn * 4 + 0];
    const float cy = boxes[(size_t)bn * 4 + 1];
    float v = (band == 0) ? cx : (band == 1) ? cy : cx * cy;
    if (band == 2) f *= 0.1f;
    const float ang = v * f;
    const float c = __cosf(ang), s = __sinf(ang);
    r0 = a0 * c - a1 * s;            // a*cos + rotate_half(a)*sin
    r1 = a1 * c + a0 * s;
  }
  qk_h[(size_t)bn * kD + d0] = (half_t)r0;
  qk_h[(size_t)bn * kD + d1] = (half_t)r1;
}

// =======================================================================
// Generic WMMA GEMM:  C[M,N] = A[M,K] @ B[K,N] + bias  (opt ReLU)
// block tile 128x128, 8 waves (4M x 2N), each wave 32x64 (2x4 wmma tiles)
// f16 A is staged to LDS via async global->LDS B128 copies (ASYNCcnt);
// f32 A / f32 weights are converted to f16 while staging through VGPRs.
// =======================================================================
template <typename AT, bool RELU, bool OUT_HALF>
__global__ __launch_bounds__(256) void gemm_wmma_kernel(
    const AT* __restrict__ A, const float* __restrict__ Bw,
    const float* __restrict__ bias, void* __restrict__ Cout,
    int M, int N, int K)
{
  constexpr int BM = 128, BN = 128, BK = 32;
  constexpr int LDA = BK + 8;
  constexpr int LDB = BK + 8;
  __shared__ alignas(16) half_t As [BM * LDA];
  __shared__ alignas(16) half_t BsT[BN * LDB];   // transposed: [col][k]

  const int tid   = threadIdx.x;
  const int lane  = tid & 31;
  const int wid   = tid >> 5;
  const int waveN = wid & 1;
  const int waveM = wid >> 1;
  const int lr    = lane & 15;
  const int lh    = lane >> 4;
  const int bm    = blockIdx.x * BM;
  const int bn    = blockIdx.y * BN;

  v8f acc[2][4] = {};

  for (int k0 = 0; k0 < K; k0 += BK) {
    {   // stage A 128x32: each thread owns 16 contiguous elements
      const int r = tid >> 1;
      const int c = (tid & 1) * 16;
      const int gr = bm + r;
      half_t* dst = &As[r * LDA + c];
      if constexpr (sizeof(AT) == 2) {
        if (gr < M) {
          const AT* src = A + (size_t)gr * K + (k0 + c);
          async_copy_b128((const half_t*)src,     dst);
          async_copy_b128((const half_t*)src + 8, dst + 8);
        } else {
#pragma unroll
          for (int i = 0; i < 16; ++i) dst[i] = (half_t)0.0f;
        }
      } else {
        half_t tmp[16];
        if (gr < M) {
          const AT* src = A + (size_t)gr * K + (k0 + c);
#pragma unroll
          for (int i = 0; i < 16; ++i) tmp[i] = (half_t)src[i];
        } else {
#pragma unroll
          for (int i = 0; i < 16; ++i) tmp[i] = (half_t)0.0f;
        }
#pragma unroll
        for (int i = 0; i < 16; ++i) dst[i] = tmp[i];
      }
    }
    {   // stage B 32x128 transposed (f32 weights -> f16)
      const int r  = tid >> 3;             // k row 0..31
      const int cb = (tid & 7) * 16;       // col base
      const float* src = Bw + (size_t)(k0 + r) * N + (bn + cb);
      if (k0 + BK < K) __builtin_prefetch(src + (size_t)BK * N, 0, 1);
#pragma unroll
      for (int i = 0; i < 16; ++i) BsT[(cb + i) * LDB + r] = (half_t)src[i];
    }
    if constexpr (sizeof(AT) == 2) wait_asynccnt0();
    __syncthreads();

    v16h af[2], bf[4];
#pragma unroll
    for (int m = 0; m < 2; ++m) {
      const half_t* p = &As[(waveM * 32 + m * 16 + lr) * LDA];
      af[m] = cat8(*(const v8h*)(p + lh * 8), *(const v8h*)(p + 16 + lh * 8));
    }
#pragma unroll
    for (int n = 0; n < 4; ++n) {
      const half_t* p = &BsT[(waveN * 64 + n * 16 + lr) * LDB];
      bf[n] = cat8(*(const v8h*)(p + lh * 8), *(const v8h*)(p + 16 + lh * 8));
    }
#pragma unroll
    for (int m = 0; m < 2; ++m)
#pragma unroll
      for (int n = 0; n < 4; ++n)
        acc[m][n] = __builtin_amdgcn_wmma_f32_16x16x32_f16(
            false, af[m], false, bf[n], (short)0, acc[m][n], false, false);
    __syncthreads();
  }

#pragma unroll
  for (int m = 0; m < 2; ++m) {
    const int row0 = bm + waveM * 32 + m * 16 + 8 * lh;
#pragma unroll
    for (int n = 0; n < 4; ++n) {
      const int col = bn + waveN * 64 + n * 16 + lr;
      const float bv = bias[col];
#pragma unroll
      for (int i = 0; i < 8; ++i) {
        const int row = row0 + i;
        if (row < M) {
          float v = acc[m][n][i] + bv;
          if (RELU) v = fmaxf(v, 0.0f);
          if constexpr (OUT_HALF)
            ((half_t*)Cout)[(size_t)row * N + col] = (half_t)v;
          else
            ((float*)Cout)[(size_t)row * N + col] = v;
        }
      }
    }
  }
}

// =======================================================================
// Self-attention (flash-style).  grid = (ceil(N/16), B*H), 1 wave/block.
// =======================================================================
__global__ __launch_bounds__(32) void attn_kernel(
    const half_t* __restrict__ Qp, const half_t* __restrict__ Kp,
    const half_t* __restrict__ Vp, half_t* __restrict__ Out)
{
  __shared__ alignas(16) half_t Pl[16 * 40];
  const int qt = blockIdx.x;
  const int b  = blockIdx.y >> 3;
  const int h  = blockIdx.y & 7;
  const int lane = threadIdx.x;
  const int lr = lane & 15, lh = lane >> 4;
  const size_t base = ((size_t)b * kN) * kD + h * kDH;

  v16h qf;
  {
    const int q = qt * 16 + lr;
    if (q < kN) {
      const half_t* p = Qp + base + (size_t)q * kD;
      qf = cat8(*(const v8h*)(p + lh * 16), *(const v8h*)(p + lh * 16 + 8));
    } else {
      half_t z[16];
#pragma unroll
      for (int i = 0; i < 16; ++i) z[i] = (half_t)0.0f;
      qf = pack16(z);
    }
  }

  float m8[8], l8[8];
  v8f o0 = {}, o1 = {};
#pragma unroll
  for (int i = 0; i < 8; ++i) { m8[i] = -1e30f; l8[i] = 0.0f; }
  const float scale = 0.17677669529663687f;   // 1/sqrt(32)

  for (int kb = 0; kb < kN; kb += 32) {
    // --- K^T fragments (B layout: col = key, k = feature) ---
    v16h kf[2];
#pragma unroll
    for (int t = 0; t < 2; ++t) {
      const int key = kb + t * 16 + lr;
      if (key < kN) {
        const half_t* p = Kp + base + (size_t)key * kD + lh * 16;
        kf[t] = cat8(*(const v8h*)p, *(const v8h*)(p + 8));
      } else {
        half_t z[16];
#pragma unroll
        for (int i = 0; i < 16; ++i) z[i] = (half_t)0.0f;
        kf[t] = pack16(z);
      }
    }
    v8f zc = {};
    v8f s0 = __builtin_amdgcn_wmma_f32_16x16x32_f16(false, qf, false, kf[0],
                                                    (short)0, zc, false, false);
    v8f s1 = __builtin_amdgcn_wmma_f32_16x16x32_f16(false, qf, false, kf[1],
                                                    (short)0, zc, false, false);
    const bool bad0 = (kb + lr)      >= kN;
    const bool bad1 = (kb + 16 + lr) >= kN;
#pragma unroll
    for (int i = 0; i < 8; ++i) {
      float x0 = s0[i] * scale; if (bad0) x0 = -1e30f;
      float x1 = s1[i] * scale; if (bad1) x1 = -1e30f;
      s0[i] = x0; s1[i] = x1;
    }
    // --- row max across the 16 key columns (16-lane halves) ---
    float mc[8];
#pragma unroll
    for (int i = 0; i < 8; ++i) mc[i] = fmaxf(s0[i], s1[i]);
#pragma unroll
    for (int off = 1; off < 16; off <<= 1)
#pragma unroll
      for (int i = 0; i < 8; ++i) mc[i] = fmaxf(mc[i], __shfl_xor(mc[i], off));

    float ps[8];
#pragma unroll
    for (int i = 0; i < 8; ++i) {
      const float mn = fmaxf(m8[i], mc[i]);
      const float r  = __expf(m8[i] - mn);
      m8[i] = mn;
      l8[i] *= r;
      o0[i] *= r;  o1[i] *= r;
      const float p0 = __expf(s0[i] - mn);
      const float p1 = __expf(s1[i] - mn);
      s0[i] = p0;  s1[i] = p1;
      ps[i] = p0 + p1;
    }
#pragma unroll
    for (int off = 1; off < 16; off <<= 1)
#pragma unroll
      for (int i = 0; i < 8; ++i) ps[i] += __shfl_xor(ps[i], off);
#pragma unroll
    for (int i = 0; i < 8; ++i) l8[i] += ps[i];

    // --- transpose P (C layout) -> A layout through LDS ---
    __syncthreads();
#pragma unroll
    for (int i = 0; i < 8; ++i) {
      Pl[(i + 8 * lh) * 40 + lr]      = (half_t)s0[i];
      Pl[(i + 8 * lh) * 40 + 16 + lr] = (half_t)s1[i];
    }
    __syncthreads();
    v16h pf;
    {
      const half_t* p = &Pl[lr * 40];
      pf = cat8(*(const v8h*)(p + lh * 16), *(const v8h*)(p + lh * 16 + 8));
    }
    // --- V fragments (B layout: col = dh, k = key) ---
    v16h vf0, vf1;
    {
      half_t t0[16], t1[16];
#pragma unroll
      for (int j = 0; j < 16; ++j) {
        const int key = kb + lh * 16 + j;
        if (key < kN) {
          const half_t* vp = Vp + base + (size_t)key * kD;
          t0[j] = vp[lr];
          t1[j] = vp[16 + lr];
        } else { t0[j] = (half_t)0.0f; t1[j] = (half_t)0.0f; }
      }
      vf0 = pack16(t0); vf1 = pack16(t1);
    }
    o0 = __builtin_amdgcn_wmma_f32_16x16x32_f16(false, pf, false, vf0,
                                                (short)0, o0, false, false);
    o1 = __builtin_amdgcn_wmma_f32_16x16x32_f16(false, pf, false, vf1,
                                                (short)0, o1, false, false);
  }

#pragma unroll
  for (int i = 0; i < 8; ++i) {
    const int row = qt * 16 + i + 8 * lh;
    if (row < kN) {
      const float inv = 1.0f / l8[i];
      Out[base + (size_t)row * kD + lr]      = (half_t)(o0[i] * inv);
      Out[base + (size_t)row * kD + 16 + lr] = (half_t)(o1[i] * inv);
    }
  }
}

// =======================================================================
// aw softmax over 16 (per b,n,h).  one thread each.
// =======================================================================
__global__ __launch_bounds__(256) void aw_softmax_kernel(
    const float* __restrict__ logits, float* __restrict__ out)
{
  const int t = blockIdx.x * 256 + threadIdx.x;     // (b*N+n)*8 + h
  if (t >= kRows * kH) return;
  const float* in = logits + (size_t)t * 16;
  float mx = -1e30f;
#pragma unroll
  for (int i = 0; i < 16; ++i) mx = fmaxf(mx, in[i]);
  float e[16], s = 0.0f;
#pragma unroll
  for (int i = 0; i < 16; ++i) { e[i] = __expf(in[i] - mx); s += e[i]; }
  const float inv = 1.0f / s;
  float* o = out + (size_t)t * 16;
#pragma unroll
  for (int i = 0; i < 16; ++i) o[i] = e[i] * inv;
}

// =======================================================================
// Deformable bilinear sampling. grid = B*N, 256 threads (= h*32+d).
// =======================================================================
__global__ __launch_bounds__(256) void deform_sample_kernel(
    const half_t* __restrict__ value, const float* __restrict__ off,
    const float* __restrict__ aw, const float* __restrict__ refp,
    half_t* __restrict__ out)
{
  const int bn = blockIdx.x;
  const int t  = threadIdx.x;
  const int h  = t >> 5, d = t & 31;
  const int b  = bn / kN;
  const float* offr = off  + (size_t)bn * 256;
  const float* awr  = aw   + (size_t)bn * 128 + h * 16;
  const float* rp   = refp + (size_t)bn * (kL * 2);
  const int HL[4] = {128, 64, 32, 16};
  const int WL[4] = {128, 64, 32, 16};
  const int ST[4] = {0, 16384, 20480, 21504};

  float accv = 0.0f;
#pragma unroll
  for (int l = 0; l < kL; ++l) {
    const int Hl = HL[l], Wl = WL[l];
    const float rx = rp[l * 2 + 0], ry = rp[l * 2 + 1];
    const half_t* vbase = value + ((size_t)b * kS + ST[l]) * kD + h * kDH + d;
#pragma unroll
    for (int p = 0; p < kP; ++p) {
      const int oi = ((h * kL + l) * kP + p) * 2;
      const float lx = rx + offr[oi]     / (float)Wl;
      const float ly = ry + offr[oi + 1] / (float)Hl;
      const float w  = awr[l * kP + p];
      const float x = lx * Wl - 0.5f;
      const float y = ly * Hl - 0.5f;
      const float x0f = floorf(x), y0f = floorf(y);
      const int x0 = (int)x0f, y0 = (int)y0f;
      const float fx = x - x0f, fy = y - y0f;
      auto tap = [&](int yi, int xi) -> float {
        if (xi < 0 || xi >= Wl || yi < 0 || yi >= Hl) return 0.0f;
        return (float)vbase[((size_t)yi * Wl + xi) * kD];
      };
      const float v00 = tap(y0,     x0);
      const float v01 = tap(y0,     x0 + 1);
      const float v10 = tap(y0 + 1, x0);
      const float v11 = tap(y0 + 1, x0 + 1);
      accv += w * (v00 * (1.f - fy) * (1.f - fx) + v01 * (1.f - fy) * fx +
                   v10 * fy * (1.f - fx)         + v11 * fy * fx);
    }
  }
  out[(size_t)bn * kD + t] = (half_t)accv;
}

// =======================================================================
// Residual add + LayerNorm. 8 waves/block, one row (D=256) per wave.
// Optionally emits f16 copy and f16 (y + extra) [for q2].
// =======================================================================
__global__ __launch_bounds__(256) void add_ln_kernel(
    const float* __restrict__ A, const float* __restrict__ Bres,
    const float* __restrict__ g, const float* __restrict__ beta,
    const float* __restrict__ extra,
    float* __restrict__ out32, half_t* __restrict__ out16,
    half_t* __restrict__ q2out, int rows)
{
  const int lane = threadIdx.x & 31;
  const int row  = blockIdx.x * 8 + (threadIdx.x >> 5);
  if (row >= rows) return;
  const float* a  = A    + (size_t)row * kD;
  const float* br = Bres + (size_t)row * kD;
  float x[8], s = 0.0f;
#pragma unroll
  for (int i = 0; i < 8; ++i) { x[i] = a[lane + i * 32] + br[lane + i * 32]; s += x[i]; }
#pragma unroll
  for (int off = 16; off >= 1; off >>= 1) s += __shfl_xor(s, off);
  const float mu = s * (1.0f / kD);
  float vs = 0.0f;
#pragma unroll
  for (int i = 0; i < 8; ++i) { const float dd = x[i] - mu; vs += dd * dd; }
#pragma unroll
  for (int off = 16; off >= 1; off >>= 1) vs += __shfl_xor(vs, off);
  const float rinv = rsqrtf(vs * (1.0f / kD) + 1e-5f);
#pragma unroll
  for (int i = 0; i < 8; ++i) {
    const int c = lane + i * 32;
    const float y = (x[i] - mu) * rinv * g[c] + beta[c];
    if (out32) out32[(size_t)row * kD + c] = y;
    if (out16) out16[(size_t)row * kD + c] = (half_t)y;
    if (q2out) q2out[(size_t)row * kD + c] = (half_t)(y + extra[(size_t)row * kD + c]);
  }
}

// =======================================================================
// Host-side orchestration
// =======================================================================
extern "C" void kernel_launch(void* const* d_in, const int* in_sizes, int n_in,
                              void* d_out, int out_size, void* d_ws, size_t ws_size,
                              hipStream_t stream)
{
  const float* tgt    = (const float*)d_in[0];
  const float* memory = (const float*)d_in[1];
  const float* qpos   = (const float*)d_in[2];
  const float* boxes  = (const float*)d_in[3];
  const float* refp   = (const float*)d_in[4];
  const float* wq     = (const float*)d_in[5];
  const float* bq     = (const float*)d_in[6];
  const float* wk     = (const float*)d_in[7];
  const float* bk     = (const float*)d_in[8];
  const float* wv     = (const float*)d_in[9];
  const float* bv     = (const float*)d_in[10];
  const float* sa_wo  = (const float*)d_in[11];
  const float* sa_bo  = (const float*)d_in[12];
  const float* n1_g   = (const float*)d_in[13];
  const float* n1_b   = (const float*)d_in[14];
  const float* off_w  = (const float*)d_in[15];
  const float* off_b  = (const float*)d_in[16];
  const float* aw_w   = (const float*)d_in[17];
  const float* aw_b   = (const float*)d_in[18];
  const float* val_w  = (const float*)d_in[19];
  const float* val_b  = (const float*)d_in[20];
  const float* co_w   = (const float*)d_in[21];
  const float* co_b   = (const float*)d_in[22];
  const float* n2_g   = (const float*)d_in[23];
  const float* n2_b   = (const float*)d_in[24];
  const float* ff1_w  = (const float*)d_in[25];
  const float* ff1_b  = (const float*)d_in[26];
  const float* ff2_w  = (const float*)d_in[27];
  const float* ff2_b  = (const float*)d_in[28];
  const float* n3_g   = (const float*)d_in[29];
  const float* n3_b   = (const float*)d_in[30];
  float* out = (float*)d_out;

  char* ws = (char*)d_ws;
  auto carve = [&](size_t bytes) -> void* {
    void* p = (void*)ws;
    ws += (bytes + 255) & ~(size_t)255;
    return p;
  };
  const size_t ROWD = (size_t)kRows * kD;                 // 7200*256
  half_t* qk_h    = (half_t*)carve(ROWD * 2);
  half_t* tgt_h   = (half_t*)carve(ROWD * 2);
  half_t* Qp      = (half_t*)carve(ROWD * 2);
  half_t* Kp      = (half_t*)carve(ROWD * 2);
  half_t* Vp      = (half_t*)carve(ROWD * 2);
  half_t* attn_o  = (half_t*)carve(ROWD * 2);
  float*  sa_proj = (float*) carve(ROWD * 4);
  float*  tgt1    = (float*) carve(ROWD * 4);
  half_t* q2_h    = (half_t*)carve(ROWD * 2);
  half_t* value_h = (half_t*)carve((size_t)kMemRows * kD * 2);
  float*  off_o   = (float*) carve(ROWD * 4);
  float*  awl     = (float*) carve((size_t)kRows * 128 * 4);
  float*  aws     = (float*) carve((size_t)kRows * 128 * 4);
  half_t* ca_s    = (half_t*)carve(ROWD * 2);
  float*  ca_p    = (float*) carve(ROWD * 4);
  float*  tgt2    = (float*) carve(ROWD * 4);
  half_t* tgt2_h  = (half_t*)carve(ROWD * 2);
  half_t* ff1_h   = (half_t*)carve((size_t)kRows * kFF * 2);
  float*  ff2_o   = (float*) carve(ROWD * 4);

  // 1) RoPE prep
  rope_kernel<<<kRows, 128, 0, stream>>>(tgt, qpos, boxes, qk_h, tgt_h);

  // 2) q/k/v projections (f16 out for attention)
  dim3 gQ((kRows + 127) / 128, kD / 128);
  gemm_wmma_kernel<half_t, false, true><<<gQ, 256, 0, stream>>>(qk_h,  wq, bq, Qp, kRows, kD, kD);
  gemm_wmma_kernel<half_t, false, true><<<gQ, 256, 0, stream>>>(qk_h,  wk, bk, Kp, kRows, kD, kD);
  gemm_wmma_kernel<half_t, false, true><<<gQ, 256, 0, stream>>>(tgt_h, wv, bv, Vp, kRows, kD, kD);

  // 3) value projection over the whole memory (A read as f32, converted in LDS)
  dim3 gV((kMemRows + 127) / 128, kD / 128);
  gemm_wmma_kernel<float, false, true><<<gV, 256, 0, stream>>>(memory, val_w, val_b, value_h,
                                                               kMemRows, kD, kD);

  // 4) self-attention
  attn_kernel<<<dim3((kN + 15) / 16, kB * kH), 32, 0, stream>>>(Qp, Kp, Vp, attn_o);

  // 5) output projection + LN1 (also produces q2 = ln1 + query_pos in f16)
  gemm_wmma_kernel<half_t, false, false><<<gQ, 256, 0, stream>>>(attn_o, sa_wo, sa_bo, sa_proj,
                                                                 kRows, kD, kD);
  add_ln_kernel<<<(kRows + 7) / 8, 256, 0, stream>>>(tgt, sa_proj, n1_g, n1_b, qpos,
                                                     tgt1, (half_t*)nullptr, q2_h, kRows);

  // 6) offsets + attention-weight logits
  gemm_wmma_kernel<half_t, false, false><<<gQ, 256, 0, stream>>>(q2_h, off_w, off_b, off_o,
                                                                 kRows, kD, kD);
  dim3 gA((kRows + 127) / 128, 128 / 128);
  gemm_wmma_kernel<half_t, false, false><<<gA, 256, 0, stream>>>(q2_h, aw_w, aw_b, awl,
                                                                 kRows, 128, kD);
  aw_softmax_kernel<<<(kRows * kH + 255) / 256, 256, 0, stream>>>(awl, aws);

  // 7) deformable sampling + output projection + LN2
  deform_sample_kernel<<<kRows, 256, 0, stream>>>(value_h, off_o, aws, refp, ca_s);
  gemm_wmma_kernel<half_t, false, false><<<gQ, 256, 0, stream>>>(ca_s, co_w, co_b, ca_p,
                                                                 kRows, kD, kD);
  add_ln_kernel<<<(kRows + 7) / 8, 256, 0, stream>>>(tgt1, ca_p, n2_g, n2_b, (const float*)nullptr,
                                                     tgt2, tgt2_h, (half_t*)nullptr, kRows);

  // 8) FFN (ReLU fused into ff1) + LN3 -> d_out
  dim3 gF1((kRows + 127) / 128, kFF / 128);
  gemm_wmma_kernel<half_t, true, true><<<gF1, 256, 0, stream>>>(tgt2_h, ff1_w, ff1_b, ff1_h,
                                                                kRows, kFF, kD);
  gemm_wmma_kernel<half_t, false, false><<<gQ, 256, 0, stream>>>(ff1_h, ff2_w, ff2_b, ff2_o,
                                                                 kRows, kD, kFF);
  add_ln_kernel<<<(kRows + 7) / 8, 256, 0, stream>>>(tgt2, ff2_o, n3_g, n3_b, (const float*)nullptr,
                                                     out, (half_t*)nullptr, (half_t*)nullptr, kRows);
}
